// ParallelLinear2D_44856638439902
// MI455X (gfx1250) — matrix operand order repaired and checked by reading, not compiled, not run on hardware
//
#include <hip/hip_runtime.h>
#include <stdint.h>

// ---------------------------------------------------------------------------
// TP linear (single device): out[M,N] = X[M,K] @ W[N,K]^T, fp32 in/out.
// M=8192, N=4096, K=4096. Compute-bound (AI ~820 FLOP/B would need ~19 PFLOPS
// to be HBM-bound), so the matrix-core path dominates: fp32 -> bf16 hi/lo
// split, 3x v_wmma_f32_16x16x32_bf16 per 16x16x32 step (hh + hl + lh), f32 acc.
// Fast path: pre-split X/W into d_ws (~200MB ~= L2-resident), then a GEMM whose
// staging uses gfx1250 async global->LDS copies (ASYNCcnt) with double-buffered
// *static* LDS and compile-time buffer selection (K-loop unrolled by 2) so the
// fragment loads stay on the ds_load path.
// ---------------------------------------------------------------------------

typedef __attribute__((ext_vector_type(16))) __bf16 v16bf;
typedef __attribute__((ext_vector_type(8)))  float  v8f;

#define TILE_M 128
#define TILE_N 128
#define TILE_K 64
#define LDT    72          // padded LDS row stride in bf16 elems (36 banks)
#define THREADS 256
#define TILE_ELEMS (TILE_M * LDT)     // one padded 128x64 tile in LDS

// ---------------- shared fragment loaders (LDS -> VGPR, wave32 WMMA layout) --

// A fragment (16x32, 16-bit): lane<16 -> M=lane, K chunks [kk,+8) & [kk+16,+8)
//                             lane>=16 -> M=lane-16, chunks [kk+8,+8) & [kk+24,+8)
__device__ __forceinline__ v16bf load_a_frag(const __bf16* s, int row, int kk, int lane)
{
  const int r     = row + (lane & 15);
  const int kbase = kk + ((lane & 16) ? 8 : 0);
  const __bf16* p = s + r * LDT + kbase;
  union { uint4 u[2]; v16bf v; } f;
  f.u[0] = *(const uint4*)(p);
  f.u[1] = *(const uint4*)(p + 16);
  return f.v;
}

// B fragment (32x16, 16-bit): lane<16 -> N=lane, K chunks [kk,+8) & [kk+16,+8);
// lane>=16 -> N=lane-16, chunks [kk+8,+8) & [kk+24,+8). W is [N,K] row-major.
__device__ __forceinline__ v16bf load_b_frag(const __bf16* s, int nrow, int kk, int lane)
{
  const int r     = nrow + (lane & 15);
  const int kbase = kk + ((lane & 16) ? 8 : 0);
  const __bf16* p = s + r * LDT + kbase;
  union { uint4 u[2]; v16bf v; } f;
  f.u[0] = *(const uint4*)(p);
  f.u[1] = *(const uint4*)(p + 16);
  return f.v;
}

// Inner 16x16x(2x32) product block shared by both kernels.
__device__ __forceinline__ void mma_block(
    const __bf16* sAhi, const __bf16* sAlo,
    const __bf16* sBhi, const __bf16* sBlo,
    int waveM, int waveN, int lane, v8f acc[4][2])
{
#pragma unroll
  for (int kk = 0; kk < TILE_K; kk += 32) {
    v16bf ah[4], al[4];
#pragma unroll
    for (int mt = 0; mt < 4; ++mt) {
      const int r = waveM * 64 + mt * 16;
      ah[mt] = load_a_frag(sAhi, r, kk, lane);
      al[mt] = load_a_frag(sAlo, r, kk, lane);
    }
#pragma unroll
    for (int nt = 0; nt < 2; ++nt) {
      const int n = waveN * 32 + nt * 16;
      const v16bf bh = load_b_frag(sBhi, n, kk, lane);
      const v16bf bl = load_b_frag(sBlo, n, kk, lane);
#pragma unroll
      for (int mt = 0; mt < 4; ++mt) {
        acc[mt][nt] = __builtin_amdgcn_wmma_f32_16x16x32_bf16(
            false, ah[mt], false, bh, (short)0, acc[mt][nt], false, false);
        acc[mt][nt] = __builtin_amdgcn_wmma_f32_16x16x32_bf16(
            false, ah[mt], false, bl, (short)0, acc[mt][nt], false, false);
        acc[mt][nt] = __builtin_amdgcn_wmma_f32_16x16x32_bf16(
            false, al[mt], false, bh, (short)0, acc[mt][nt], false, false);
      }
    }
  }
}

__device__ __forceinline__ void store_epilogue(
    float* __restrict__ C, int N, int blockM, int blockN,
    int waveM, int waveN, int lane, v8f acc[4][2])
{
  const int nlane = lane & 15;
  const int mhalf = (lane & 16) ? 8 : 0;
#pragma unroll
  for (int mt = 0; mt < 4; ++mt) {
#pragma unroll
    for (int nt = 0; nt < 2; ++nt) {
      const int n = blockN + waveN * 32 + nt * 16 + nlane;
#pragma unroll
      for (int i = 0; i < 8; ++i) {
        const int m = blockM + waveM * 64 + mt * 16 + mhalf + i;
        __builtin_nontemporal_store(acc[mt][nt][i], C + (size_t)m * N + n);
      }
    }
  }
}

// ---------------- fast path: pre-split bf16 planes + async-LDS staging -------

// fp32 -> bf16 hi/lo split, vectorized float4 -> 2x uint2 (4 bf16 each).
__global__ void split_bf16_kernel(const float* __restrict__ x,
                                  __bf16* __restrict__ hi,
                                  __bf16* __restrict__ lo, long n4)
{
  const long i = (long)blockIdx.x * blockDim.x + threadIdx.x;
  if (i >= n4) return;
  const float4 v = *(const float4*)(x + i * 4);
  union { __bf16 h[4]; uint2 u; } ph, pl;
  ph.h[0] = (__bf16)v.x;  pl.h[0] = (__bf16)(v.x - (float)ph.h[0]);
  ph.h[1] = (__bf16)v.y;  pl.h[1] = (__bf16)(v.y - (float)ph.h[1]);
  ph.h[2] = (__bf16)v.z;  pl.h[2] = (__bf16)(v.z - (float)ph.h[2]);
  ph.h[3] = (__bf16)v.w;  pl.h[3] = (__bf16)(v.w - (float)ph.h[3]);
  *(uint2*)(hi + i * 4) = ph.u;
  *(uint2*)(lo + i * 4) = pl.u;
}

// Issue 4 async 16B global->LDS copies for one 128x64 bf16 tile (no VGPR
// round-trip; tracked by ASYNCcnt). LDS byte address = low 32 bits of the
// generic pointer (LDS aperture, ISA 10.2).
__device__ __forceinline__ void stage_async_bf16(
    const __bf16* __restrict__ g, int ldg, int row0, int k0, int tid,
    __bf16* s)
{
#pragma unroll
  for (int it = 0; it < 4; ++it) {
    const int chunk = tid + it * THREADS;   // 1024 chunks of 8 bf16 (16B)
    const int row   = chunk >> 3;
    const int col   = (chunk & 7) << 3;
    const __bf16* gp = g + (size_t)(row0 + row) * ldg + k0 + col;
    const uint32_t laddr = (uint32_t)(uintptr_t)(s + row * LDT + col);
    asm volatile("global_load_async_to_lds_b128 %0, %1, off"
                 :: "v"(laddr), "v"(gp) : "memory");
  }
}

__device__ __forceinline__ void wait_async_all()
{
  asm volatile("s_wait_asynccnt 0x0" ::: "memory");
}

__global__ __launch_bounds__(THREADS)
void gemm_presplit_bf16x3(
    const __bf16* __restrict__ Ahi, const __bf16* __restrict__ Alo,  // [M,K]
    const __bf16* __restrict__ Whi, const __bf16* __restrict__ Wlo,  // [N,K]
    float* __restrict__ C, int M, int N, int K)
{
  // Double-buffered static LDS: 2 buffers x 4 padded tiles = 144 KB.
  __shared__ __bf16 smem[8 * TILE_ELEMS];
#define SBUF(b, t) (smem + ((b) * 4 + (t)) * TILE_ELEMS)

  const int tid   = threadIdx.x;
  const int lane  = tid & 31;
  const int wave  = tid >> 5;
  const int waveM = wave >> 2;
  const int waveN = wave & 3;
  const int blockM = blockIdx.y * TILE_M;
  const int blockN = blockIdx.x * TILE_N;

  v8f acc[4][2] = {};

  // Prime the pipeline: async-fill buffer 0 with tile kt=0.
  stage_async_bf16(Ahi, K, blockM, 0, tid, SBUF(0, 0));
  stage_async_bf16(Alo, K, blockM, 0, tid, SBUF(0, 1));
  stage_async_bf16(Whi, K, blockN, 0, tid, SBUF(0, 2));
  stage_async_bf16(Wlo, K, blockN, 0, tid, SBUF(0, 3));

  // K-loop unrolled by 2 so buffer selection is compile-time constant
  // (keeps fragment loads on the ds_load path, no indexed pointers).
  for (int kt = 0; kt < K; kt += 2 * TILE_K) {
    // ---- phase 0: compute on buffer 0, prefetch into buffer 1 ----
    wait_async_all();        // this wave's fills of buf0 have landed
    __syncthreads();         // ... and everyone else's too

    if (kt + TILE_K < K) {
      stage_async_bf16(Ahi, K, blockM, kt + TILE_K, tid, SBUF(1, 0));
      stage_async_bf16(Alo, K, blockM, kt + TILE_K, tid, SBUF(1, 1));
      stage_async_bf16(Whi, K, blockN, kt + TILE_K, tid, SBUF(1, 2));
      stage_async_bf16(Wlo, K, blockN, kt + TILE_K, tid, SBUF(1, 3));
    }
    mma_block(SBUF(0, 0), SBUF(0, 1), SBUF(0, 2), SBUF(0, 3),
              waveM, waveN, lane, acc);

    // ---- phase 1: compute on buffer 1, prefetch into buffer 0 ----
    if (kt + TILE_K < K) {
      wait_async_all();
      __syncthreads();

      if (kt + 2 * TILE_K < K) {
        stage_async_bf16(Ahi, K, blockM, kt + 2 * TILE_K, tid, SBUF(0, 0));
        stage_async_bf16(Alo, K, blockM, kt + 2 * TILE_K, tid, SBUF(0, 1));
        stage_async_bf16(Whi, K, blockN, kt + 2 * TILE_K, tid, SBUF(0, 2));
        stage_async_bf16(Wlo, K, blockN, kt + 2 * TILE_K, tid, SBUF(0, 3));
      }
      mma_block(SBUF(1, 0), SBUF(1, 1), SBUF(1, 2), SBUF(1, 3),
                waveM, waveN, lane, acc);
    }
  }
#undef SBUF

  store_epilogue(C, N, blockM, blockN, waveM, waveN, lane, acc);
}

// ---------------- fallback path: convert while staging (no workspace) -------

__device__ __forceinline__ void stage_tile_split(
    const float* __restrict__ g, int ldg, int row0, int k0,
    __bf16* __restrict__ hi, __bf16* __restrict__ lo, int tid)
{
#pragma unroll
  for (int it = 0; it < (TILE_M * TILE_K / 4) / THREADS; ++it) {
    const int chunk = tid + it * THREADS;
    const int row   = chunk >> 4;
    const int col   = (chunk & 15) << 2;
    const float4 v = *(const float4*)(g + (size_t)(row0 + row) * ldg + k0 + col);
    union { __bf16 h[4]; uint2 u; } ph, pl;
    ph.h[0] = (__bf16)v.x;  pl.h[0] = (__bf16)(v.x - (float)ph.h[0]);
    ph.h[1] = (__bf16)v.y;  pl.h[1] = (__bf16)(v.y - (float)ph.h[1]);
    ph.h[2] = (__bf16)v.z;  pl.h[2] = (__bf16)(v.z - (float)ph.h[2]);
    ph.h[3] = (__bf16)v.w;  pl.h[3] = (__bf16)(v.w - (float)ph.h[3]);
    *(uint2*)(hi + row * LDT + col) = ph.u;
    *(uint2*)(lo + row * LDT + col) = pl.u;
  }
}

__global__ __launch_bounds__(THREADS)
void gemm_convert_bf16x3(
    const float* __restrict__ A, const float* __restrict__ W,
    float* __restrict__ C, int M, int N, int K)
{
  __shared__ __bf16 sAhi[TILE_ELEMS];
  __shared__ __bf16 sAlo[TILE_ELEMS];
  __shared__ __bf16 sBhi[TILE_ELEMS];
  __shared__ __bf16 sBlo[TILE_ELEMS];

  const int tid   = threadIdx.x;
  const int lane  = tid & 31;
  const int wave  = tid >> 5;
  const int waveM = wave >> 2;
  const int waveN = wave & 3;
  const int blockM = blockIdx.y * TILE_M;
  const int blockN = blockIdx.x * TILE_N;

  v8f acc[4][2] = {};

  for (int kt = 0; kt < K; kt += TILE_K) {
    stage_tile_split(A, K, blockM, kt, sAhi, sAlo, tid);
    stage_tile_split(W, K, blockN, kt, sBhi, sBlo, tid);
    __syncthreads();
    mma_block(sAhi, sAlo, sBhi, sBlo, waveM, waveN, lane, acc);
    __syncthreads();
  }

  store_epilogue(C, N, blockM, blockN, waveM, waveN, lane, acc);
}

// ---------------- misc -------------------------------------------------------

__global__ void bias_passthrough(const float* __restrict__ b,
                                 float* __restrict__ o, int n)
{
  const int i = blockIdx.x * blockDim.x + threadIdx.x;
  if (i < n) o[i] = b[i];
}

extern "C" void kernel_launch(void* const* d_in, const int* in_sizes, int n_in,
                              void* d_out, int out_size, void* d_ws, size_t ws_size,
                              hipStream_t stream)
{
  const float* A    = (const float*)d_in[0];   // [S,B,IN] == [M,K]
  const float* W    = (const float*)d_in[1];   // [OUT,IN] == [N,K]
  const float* bias = (const float*)d_in[2];   // [OUT]
  float* out = (float*)d_out;

  const int N = in_sizes[2];                   // OUT
  const int K = in_sizes[1] / N;               // IN
  const int M = in_sizes[0] / K;               // S*B

  const size_t nA = (size_t)M * K;             // elems
  const size_t nW = (size_t)N * K;
  const size_t need = (nA + nW) * 2 * sizeof(__bf16);

  dim3 grid(N / TILE_N, M / TILE_M);

  if (ws_size >= need) {
    __bf16* Ahi = (__bf16*)d_ws;
    __bf16* Alo = Ahi + nA;
    __bf16* Whi = Alo + nA;
    __bf16* Wlo = Whi + nW;

    const long a4 = (long)(nA / 4), w4 = (long)(nW / 4);
    split_bf16_kernel<<<(a4 + 255) / 256, 256, 0, stream>>>(A, Ahi, Alo, a4);
    split_bf16_kernel<<<(w4 + 255) / 256, 256, 0, stream>>>(W, Whi, Wlo, w4);

    gemm_presplit_bf16x3<<<grid, THREADS, 0, stream>>>(Ahi, Alo, Whi, Wlo,
                                                       out, M, N, K);
  } else {
    gemm_convert_bf16x3<<<grid, THREADS, 0, stream>>>(A, W, out, M, N, K);
  }

  // Tuple output: bias returned unchanged (skip_bias_add), appended after GEMM.
  float* bias_out = out + (size_t)M * N;
  bias_passthrough<<<(N + 255) / 256, 256, 0, stream>>>(bias, bias_out, N);
}
